// F2NetHead_27530740367627
// MI455X (gfx1250) — matrix operand order repaired and checked by reading, not compiled, not run on hardware
//
#include <hip/hip_runtime.h>
#include <cstdint>
#include <cstddef>

// ---------------------------------------------------------------------------
// F2NetHead for MI455X (gfx1250, wave32, WMMA).
// All four GEMMs run on v_wmma_f32_16x16x32_bf16 (f32 accumulate) with
// double-buffered LDS fed by global_load_async_to_lds_b128 (ASYNCcnt path).
// Conv1d(d,d,3) is folded into a K=3072 GEMM over shift-gathered Q.
// softmax(axis=-2)+glob fused into one column reduction (attn never stored).
// cumsum + silu fused into a per-column blocked scan.
// ---------------------------------------------------------------------------

typedef __bf16 bf16;
typedef __attribute__((ext_vector_type(16))) __bf16 v16bf;
typedef __attribute__((ext_vector_type(8)))  float  v8f;

#define TILE_M 128
#define TILE_N 128
#define TILE_K 32
#define LDSS   48   // padded LDS row stride in bf16 (96 bytes, keeps b128 ops aligned)

// ---- CDNA5 async global->LDS copy (16B per lane, tracked by ASYNCcnt) -----
__device__ __forceinline__ void async_copy_b128(void* lds_dst, const void* gsrc) {
    // LDS aperture keeps the wave-relative LDS offset in addr[31:0] (ISA 10.2)
    unsigned lds = (unsigned)(uintptr_t)lds_dst;
    unsigned long long ga = (unsigned long long)(uintptr_t)gsrc;
    asm volatile("global_load_async_to_lds_b128 %0, %1, off"
                 :: "v"(lds), "v"(ga) : "memory");
}
__device__ __forceinline__ void wait_async0() {
    asm volatile("s_wait_asynccnt 0x0" ::: "memory");
}

// ----------------------------- small helpers -------------------------------

__global__ void k_f32_to_bf16(const float* __restrict__ in, bf16* __restrict__ out, size_t n) {
    size_t i = (size_t)blockIdx.x * blockDim.x + threadIdx.x;
    if (i < n) out[i] = (bf16)in[i];
}

// conv_w [O=1024][I=1024][K=3] (row-major) -> Wc [O][j*1024 + i] bf16
__global__ void k_pack_conv(const float* __restrict__ in, bf16* __restrict__ out, size_t n) {
    size_t idx = (size_t)blockIdx.x * blockDim.x + threadIdx.x;
    if (idx >= n) return;
    int o = (int)(idx / 3072);
    int t = (int)(idx % 3072);
    int j = t >> 10;         // 0..2
    int i = t & 1023;
    out[idx] = (bf16)in[(size_t)o * 3072 + (size_t)i * 3 + j];
}

// ----------------------------- WMMA GEMM -----------------------------------
// out[m,n] = epilogue( sum_k A[m,k] * Bw[n,k] )   (Bw row-major [N][K])
// MODE 0: qog   : +bias, split -> Qb(bf16), O(f32), G(f32)   (N=3072)
// MODE 1: conv  : A = Q with per-K-block row shift j-1, +bias, silu -> bf16
// MODE 2: scores: * scale -> f32
// MODE 3: out   : +bias -> f32 (d_out)
template <int MODE>
__global__ __launch_bounds__(256) void k_wmma_gemm(
    const bf16* __restrict__ A, const bf16* __restrict__ Bw,
    const float* __restrict__ bias,
    void* __restrict__ out0, void* __restrict__ out1, void* __restrict__ out2,
    int Mdim, int Ndim, int Ktot, int Sdim, float scale)
{
    __shared__ bf16 lsA[2][TILE_M * LDSS];
    __shared__ bf16 lsB[2][TILE_N * LDSS];

    const int tid  = threadIdx.x;
    const int lane = tid & 31;
    const int half = lane >> 4;    // ISA 7.12.2 half-wave selector
    const int ln   = lane & 15;
    const int wv   = tid >> 5;     // 8 waves
    const int wm   = wv & 3;       // 4 waves along M (32 rows each)
    const int wn   = wv >> 2;      // 2 waves along N (64 cols each)

    const long blockM = (long)blockIdx.y * TILE_M;
    const int  blockN = blockIdx.x * TILE_N;
    const int  ldA    = (MODE == 1) ? 1024 : Ktot;

    // issue one 128x32 A tile + 128x32 B tile into LDS buffer `buf`
    auto issue_tile = [&](int k0, int buf) {
        int jsh = 0, i0 = k0;
        if (MODE == 1) { jsh = (k0 >> 10) - 1; i0 = k0 & 1023; }
#pragma unroll
        for (int t = 0; t < 2; ++t) {
            int chunk = tid + t * 256;        // 0..511
            int row   = chunk >> 2;           // 0..127
            int cc    = (chunk & 3) * 8;      // bf16 col offset {0,8,16,24}

            // A tile (with optional conv shift, zero-padded at batch edges)
            long gr = blockM + row;
            long sr = gr;
            bool valid = true;
            if (MODE == 1) {
                int s  = (int)(gr % Sdim);
                int s2 = s + jsh;
                valid  = (s2 >= 0) && (s2 < Sdim);
                sr     = gr + jsh;
            }
            if (valid) {
                async_copy_b128(&lsA[buf][row * LDSS + cc], A + sr * (long)ldA + i0 + cc);
            } else {
                *reinterpret_cast<uint4*>(&lsA[buf][row * LDSS + cc]) = make_uint4(0u, 0u, 0u, 0u);
            }

            // B tile
            long gn = (long)blockN + row;
            async_copy_b128(&lsB[buf][row * LDSS + cc], Bw + gn * (long)Ktot + k0 + cc);
        }
    };

    v8f acc[2][4];
#pragma unroll
    for (int i = 0; i < 2; ++i)
#pragma unroll
        for (int j = 0; j < 4; ++j) acc[i][j] = (v8f){0.f,0.f,0.f,0.f,0.f,0.f,0.f,0.f};

    issue_tile(0, 0);
    int cur = 0;

    for (int k0 = 0; k0 < Ktot; k0 += TILE_K) {
        wait_async0();           // own async copies into buf `cur` landed
        __syncthreads();         // all waves' copies landed; prev reads of buf cur^1 done
        if (k0 + TILE_K < Ktot)
            issue_tile(k0 + TILE_K, cur ^ 1);

        // ---- fragment gather per ISA 16-bit A 16x32 / B 32x16 layouts
        v16bf afr[2], bfr[4];
#pragma unroll
        for (int i = 0; i < 2; ++i) {
            const bf16* p = &lsA[cur][(wm * 32 + i * 16 + ln) * LDSS];
#pragma unroll
            for (int q = 0; q < 8; ++q) {
                int kk = (q < 4) ? (half * 8 + q * 2) : (16 + half * 8 + (q - 4) * 2);
                afr[i][2 * q]     = p[kk];
                afr[i][2 * q + 1] = p[kk + 1];
            }
        }
#pragma unroll
        for (int j = 0; j < 4; ++j) {
            const bf16* p = &lsB[cur][(wn * 64 + j * 16 + ln) * LDSS];
#pragma unroll
            for (int q = 0; q < 8; ++q) {
                int kk = (q < 4) ? (half * 8 + q * 2) : (16 + half * 8 + (q - 4) * 2);
                bfr[j][2 * q]     = p[kk];
                bfr[j][2 * q + 1] = p[kk + 1];
            }
        }

        // ---- 8 WMMAs per K step
#pragma unroll
        for (int i = 0; i < 2; ++i)
#pragma unroll
            for (int j = 0; j < 4; ++j)
                acc[i][j] = __builtin_amdgcn_wmma_f32_16x16x32_bf16(
                    false, afr[i], false, bfr[j], (short)0, acc[i][j], false, false);

        cur ^= 1;
    }

    // ---- epilogue (C/D layout: lane n = ln, M = half*8 + r)
#pragma unroll
    for (int i = 0; i < 2; ++i)
#pragma unroll
        for (int j = 0; j < 4; ++j) {
            long gm0 = blockM + wm * 32 + i * 16 + half * 8;
            int  gn  = blockN + wn * 64 + j * 16 + ln;
            float bv = (MODE == 2) ? 0.f : bias[gn];
#pragma unroll
            for (int r = 0; r < 8; ++r) {
                long row = gm0 + r;
                float v  = acc[i][j][r];
                if (MODE == 0) {
                    v += bv;
                    if (gn < 1024)       ((bf16*)out0)[row * 1024 + gn]          = (bf16)v;
                    else if (gn < 2048)  ((float*)out1)[row * 1024 + (gn - 1024)] = v;
                    else                 ((float*)out2)[row * 1024 + (gn - 2048)] = v;
                } else if (MODE == 1) {
                    v += bv;
                    v  = v / (1.f + __expf(-v));                 // silu
                    ((bf16*)out0)[row * (long)Ndim + gn] = (bf16)v;
                } else if (MODE == 2) {
                    ((float*)out0)[row * (long)Ndim + gn] = v * scale;
                } else {
                    ((float*)out0)[row * (long)Ndim + gn] = v + bv;
                }
            }
        }
}

// ------------------- fused column softmax(axis=-2) + glob ------------------
// glob[b,d] = sum_s Q * e^{sc - max} / sum_s e^{sc - max}
__global__ __launch_bounds__(256) void k_softmax_glob(
    const float* __restrict__ scores, const bf16* __restrict__ Qb,
    float* __restrict__ glob, int Sv, int Dv)
{
    __shared__ float red[256];
    const int blk = blockIdx.x;
    const int b   = blk / Dv;
    const int d   = blk % Dv;
    const size_t base = ((size_t)b * Sv) * Dv + d;
    const int tid = threadIdx.x;

    float mx = -3.4e38f;
    for (int s = tid; s < Sv; s += 256) mx = fmaxf(mx, scores[base + (size_t)s * Dv]);
    red[tid] = mx; __syncthreads();
    for (int o = 128; o > 0; o >>= 1) { if (tid < o) red[tid] = fmaxf(red[tid], red[tid + o]); __syncthreads(); }
    mx = red[0]; __syncthreads();

    float se = 0.f, sq = 0.f;
    for (int s = tid; s < Sv; s += 256) {
        size_t idx = base + (size_t)s * Dv;
        float e = __expf(scores[idx] - mx);
        se += e;
        sq += (float)Qb[idx] * e;
    }
    red[tid] = se; __syncthreads();
    for (int o = 128; o > 0; o >>= 1) { if (tid < o) red[tid] += red[tid + o]; __syncthreads(); }
    se = red[0]; __syncthreads();
    red[tid] = sq; __syncthreads();
    for (int o = 128; o > 0; o >>= 1) { if (tid < o) red[tid] += red[tid + o]; __syncthreads(); }
    if (tid == 0) glob[(size_t)b * Dv + d] = red[0] / se;
}

// ------------- per-column scan: L = silu(G) * cumsum(O * glob) -------------
__global__ __launch_bounds__(256) void k_scan_silu(
    const float* __restrict__ Of, const float* __restrict__ Gf,
    const float* __restrict__ glob, bf16* __restrict__ Lb, int Sv, int Dv)
{
    __shared__ float sc[256];
    const int blk = blockIdx.x;
    const int b   = blk / Dv;
    const int d   = blk % Dv;
    const float g = glob[(size_t)b * Dv + d];
    const int tid = threadIdx.x;

    float carry = 0.f;
    for (int cs = 0; cs < Sv; cs += 256) {
        int s = cs + tid;
        size_t idx = ((size_t)b * Sv + s) * Dv + d;
        float v = Of[idx] * g;
        sc[tid] = v; __syncthreads();
        for (int o = 1; o < 256; o <<= 1) {               // Hillis-Steele inclusive scan
            float add = (tid >= o) ? sc[tid - o] : 0.f;
            __syncthreads();
            sc[tid] += add;
            __syncthreads();
        }
        float cum   = carry + sc[tid];
        float total = sc[255];
        float gv    = Gf[idx];
        float L     = (gv / (1.f + __expf(-gv))) * cum;   // silu(G) * cumsum
        Lb[idx]     = (bf16)L;
        carry += total;
        __syncthreads();
    }
}

// --------------------------------- host ------------------------------------

extern "C" void kernel_launch(void* const* d_in, const int* in_sizes, int n_in,
                              void* d_out, int out_size, void* d_ws, size_t ws_size,
                              hipStream_t stream) {
    const float* x      = (const float*)d_in[0];   // [4,4096,1024]
    const float* W_qog  = (const float*)d_in[1];   // [3072,1024]
    const float* b_qog  = (const float*)d_in[2];   // [3072]
    const float* conv_w = (const float*)d_in[3];   // [1024,1024,3]
    const float* conv_b = (const float*)d_in[4];   // [1024]
    const float* w_a    = (const float*)d_in[5];   // [1024,1024]
    const float* W_out  = (const float*)d_in[6];   // [1024,1024]
    const float* b_out  = (const float*)d_in[7];   // [1024]
    float* out = (float*)d_out;                    // [4,4096,1024]

    const int Bz = 4, S = 4096, D = 1024;
    const long M = (long)Bz * S;                   // 16384

    // workspace carve-out
    char* base = (char*)d_ws;
    size_t off = 0;
    auto carve = [&](size_t bytes) -> void* {
        void* p = base + off;
        off = (off + bytes + 255) & ~(size_t)255;
        return p;
    };
    bf16*  xb     = (bf16*) carve((size_t)M * D * 2);       // x bf16
    bf16*  Wqogb  = (bf16*) carve((size_t)3 * D * D * 2);   // W_qog bf16
    bf16*  Wcb    = (bf16*) carve((size_t)D * 3 * D * 2);   // packed conv weights
    bf16*  wab    = (bf16*) carve((size_t)D * D * 2);
    bf16*  Woutb  = (bf16*) carve((size_t)D * D * 2);
    bf16*  Qb     = (bf16*) carve((size_t)M * D * 2);
    float* Ofp    = (float*)carve((size_t)M * D * 4);
    float* Gfp    = (float*)carve((size_t)M * D * 4);
    bf16*  cqb    = (bf16*) carve((size_t)M * D * 2);
    float* scores = (float*)carve((size_t)M * D * 4);
    float* glob   = (float*)carve((size_t)Bz * D * 4);
    bf16*  Lb     = (bf16*) carve((size_t)M * D * 2);
    (void)ws_size; (void)in_sizes; (void)n_in; (void)out_size;

    const float scale = 0.03125f; // 1/sqrt(1024)

    // 1) precision conversion / weight repack
    {
        size_t n = (size_t)M * D;
        k_f32_to_bf16<<<dim3((unsigned)((n + 255) / 256)), dim3(256), 0, stream>>>(x, xb, n);
        n = (size_t)3 * D * D;
        k_f32_to_bf16<<<dim3((unsigned)((n + 255) / 256)), dim3(256), 0, stream>>>(W_qog, Wqogb, n);
        n = (size_t)D * D;
        k_f32_to_bf16<<<dim3((unsigned)((n + 255) / 256)), dim3(256), 0, stream>>>(w_a, wab, n);
        k_f32_to_bf16<<<dim3((unsigned)((n + 255) / 256)), dim3(256), 0, stream>>>(W_out, Woutb, n);
        n = (size_t)D * 3 * D;
        k_pack_conv<<<dim3((unsigned)((n + 255) / 256)), dim3(256), 0, stream>>>(conv_w, Wcb, n);
    }

    dim3 blk(256);
    // 2) qog GEMM: [M,1024] x [3072,1024]^T -> Qb(bf16), O(f32), G(f32)
    k_wmma_gemm<0><<<dim3(3 * D / TILE_N, (unsigned)(M / TILE_M)), blk, 0, stream>>>(
        xb, Wqogb, b_qog, Qb, Ofp, Gfp, (int)M, 3 * D, D, S, 1.f);
    // 3) conv-as-GEMM (K=3072, shifted A) + bias + silu -> cq bf16
    k_wmma_gemm<1><<<dim3(D / TILE_N, (unsigned)(M / TILE_M)), blk, 0, stream>>>(
        Qb, Wcb, conv_b, cqb, nullptr, nullptr, (int)M, D, 3 * D, S, 1.f);
    // 4) scores = (cq @ w_a^T) * scale
    k_wmma_gemm<2><<<dim3(D / TILE_N, (unsigned)(M / TILE_M)), blk, 0, stream>>>(
        cqb, wab, nullptr, scores, nullptr, nullptr, (int)M, D, D, S, scale);
    // 5) fused column softmax + glob
    k_softmax_glob<<<dim3((unsigned)(Bz * D)), blk, 0, stream>>>(scores, Qb, glob, S, D);
    // 6) L = silu(G) * cumsum(O * glob)
    k_scan_silu<<<dim3((unsigned)(Bz * D)), blk, 0, stream>>>(Ofp, Gfp, glob, Lb, S, D);
    // 7) R = L @ W_out^T + b_out -> d_out (f32)
    k_wmma_gemm<3><<<dim3(D / TILE_N, (unsigned)(M / TILE_M)), blk, 0, stream>>>(
        Lb, Woutb, b_out, out, nullptr, nullptr, (int)M, D, D, S, 1.f);
}
